// InLay_61203283968377
// MI455X (gfx1250) — matrix operand order repaired and checked by reading, not compiled, not run on hardware
//
#include <hip/hip_runtime.h>
#include <math.h>

// ---------------------------------------------------------------------------
// Types for CDNA5 WMMA (wave32): v_wmma_f32_16x16x32_bf16
// ---------------------------------------------------------------------------
typedef __bf16 v16bf __attribute__((ext_vector_type(16)));
typedef __bf16 bf16x8 __attribute__((ext_vector_type(8)));
typedef float  v8f    __attribute__((ext_vector_type(8)));

union V16U { v16bf v; bf16x8 h[2]; };

static __device__ __forceinline__ v16bf mk16(bf16x8 lo, bf16x8 hi) {
    V16U u; u.h[0] = lo; u.h[1] = hi; return u.v;
}

static __device__ __forceinline__ bf16x8 zero8() {
    bf16x8 v;
#pragma unroll
    for (int i = 0; i < 8; ++i) v[i] = (__bf16)0.0f;
    return v;
}

static __device__ __forceinline__ v8f wmma_bf16(v16bf a, v16bf b, v8f c) {
    // D = A(16x32 bf16) * B(32x16 bf16) + C(16x16 f32)
    return __builtin_amdgcn_wmma_f32_16x16x32_bf16(
        /*neg_a=*/false, a, /*neg_b=*/false, b,
        /*c_mod=*/(short)0, c, /*reuse_a=*/false, /*reuse_b=*/false);
}

// ---------------------------------------------------------------------------
// CDNA5 async global->LDS copy (ASYNCcnt path), per CDNA5 ISA.
// Each lane supplies its own 64-bit global address and 32-bit LDS offset.
// ---------------------------------------------------------------------------
static __device__ __forceinline__ void async_ld_b128(const __bf16* g, const __bf16* l) {
    unsigned       lds = (unsigned)(unsigned long long)l;         // low 32 = LDS offset
    unsigned long long ga = (unsigned long long)g;
    asm volatile("global_load_async_to_lds_b128 %0, %1, off"
                 :: "v"(lds), "v"(ga) : "memory");
}

static __device__ __forceinline__ void wait_async0() {
#if __has_builtin(__builtin_amdgcn_s_wait_asynccnt)
    __builtin_amdgcn_s_wait_asynccnt(0);
#else
    asm volatile("s_wait_asynccnt 0x0" ::: "memory");
#endif
}

// ---------------------------------------------------------------------------
// Problem constants
// ---------------------------------------------------------------------------
#define BATCH   256
#define SEQ     49
#define DMODEL  512
#define DQK     2048
#define NHEADS  32
#define HQK     64
#define HV      16
#define MTOT    (BATCH * SEQ)       // 12544

// ---------------------------------------------------------------------------
// Conversion / packing kernels (cheap, memory-light)
// ---------------------------------------------------------------------------
__global__ void cvt_bf16(const float* __restrict__ s, __bf16* __restrict__ d, int n) {
    int i = blockIdx.x * 256 + threadIdx.x;
    if (i < n) d[i] = (__bf16)s[i];
}

// src: f32 [K][N] row-major  ->  dst: bf16 [N][K] (transposed, K contiguous)
__global__ void cvt_t_bf16(const float* __restrict__ s, __bf16* __restrict__ d,
                           int N, int K) {
    long i = (long)blockIdx.x * 256 + threadIdx.x;
    if (i < (long)N * K) {
        long n = i / K, k = i % K;
        d[i] = (__bf16)s[k * N + n];
    }
}

// tokens f32 [49][512] -> bf16 [h][d][k] = [32][16][64], zero-padded k>=49
__global__ void pack_tokens(const float* __restrict__ tok, __bf16* __restrict__ d) {
    int i = blockIdx.x * 256 + threadIdx.x;         // 32768
    int h = i >> 10, rem = i & 1023, dd = rem >> 6, k = rem & 63;
    float v = (k < SEQ) ? tok[k * DMODEL + h * HV + dd] : 0.0f;
    d[i] = (__bf16)v;
}

// ---------------------------------------------------------------------------
// Tiled WMMA GEMM with async-to-LDS double buffering.
//   C = act(A[M][K] * Wt[N][K]^T + bias), bf16 in, f32 accumulate.
// BM=128, BN=128, BK=32; 256 threads = 8 waves in a 4(M) x 2(N) grid,
// each wave computes a 32x64 block as eight 16x16 WMMA accumulators.
// Tiles are fetched with GLOBAL_LOAD_ASYNC_TO_LDS_B128 into the next buffer
// while the current buffer feeds the matrix units.
// ---------------------------------------------------------------------------
#define BM   128
#define BN   128
#define BK   32
#define TSTR 40     // padded LDS row stride (elements) to spread banks

template <bool GELU, typename OutT>
__global__ __launch_bounds__(256) void gemm_bt(
    const __bf16* __restrict__ A,   // [M][K]
    const __bf16* __restrict__ Wt,  // [N][K]
    const float*  __restrict__ bias,// [N]
    OutT*         __restrict__ C,   // [M][N]
    int M, int N, int K)
{
    __shared__ __align__(16) __bf16 As[2][BM * TSTR];   // 2 x 10240 B
    __shared__ __align__(16) __bf16 Ws[2][BN * TSTR];   // 2 x 10240 B

    const int t    = threadIdx.x;
    const int lane = t & 31;
    const int wave = t >> 5;
    const int wm   = wave >> 1;      // 0..3 -> 32-row slab
    const int wn   = wave & 1;       // 0..1 -> 64-col slab
    const int ln   = lane & 15;
    const int hi   = lane >> 4;      // 0/1

    const long mb = (long)blockIdx.x * BM;
    const long nb = (long)blockIdx.y * BN;

    // staging assignment: 2 threads per row, 32 B each (2 x b128)
    const int sr  = t >> 1;          // 0..127
    const int sc8 = (t & 1) * 16;    // element offset 0 / 16

    auto issue_tile = [&](int kb, int buf) {
        const __bf16* ga = A  + (mb + sr) * K + kb + sc8;
        const __bf16* gw = Wt + (nb + sr) * K + kb + sc8;
        async_ld_b128(ga,     &As[buf][sr * TSTR + sc8]);
        async_ld_b128(ga + 8, &As[buf][sr * TSTR + sc8 + 8]);
        async_ld_b128(gw,     &Ws[buf][sr * TSTR + sc8]);
        async_ld_b128(gw + 8, &Ws[buf][sr * TSTR + sc8 + 8]);
    };

    v8f acc[2][4] = {};

    issue_tile(0, 0);
    int cur = 0;
    for (int kb = 0; kb < K; kb += BK) {
        wait_async0();          // my async writes for buffer `cur` landed
        __syncthreads();        // everyone's landed; prior reads drained
        if (kb + BK < K) issue_tile(kb + BK, cur ^ 1);

        const __bf16* Ab = &As[cur][0];
        const __bf16* Wb = &Ws[cur][0];

        // ---- fragments per ISA 16-bit A(16x32) / B(32x16) layouts ----
        v16bf aF[2], bF[4];
#pragma unroll
        for (int mi = 0; mi < 2; ++mi) {
            const int row = wm * 32 + mi * 16 + ln;
            const int off = hi * 8;  // lanes>=16 hold K=8..15 / 24..31
            bf16x8 lo = *(const bf16x8*)(Ab + row * TSTR + off);
            bf16x8 h8 = *(const bf16x8*)(Ab + row * TSTR + 16 + off);
            aF[mi] = mk16(lo, h8);
        }
#pragma unroll
        for (int ni = 0; ni < 4; ++ni) {
            const int row = wn * 64 + ni * 16 + ln;
            const int off = hi * 8;  // lanes>=16 hold K=8..15 / 24..31
            bf16x8 lo = *(const bf16x8*)(Wb + row * TSTR + off);
            bf16x8 h8 = *(const bf16x8*)(Wb + row * TSTR + 16 + off);
            bF[ni] = mk16(lo, h8);
        }
#pragma unroll
        for (int mi = 0; mi < 2; ++mi)
#pragma unroll
            for (int ni = 0; ni < 4; ++ni)
                acc[mi][ni] = wmma_bf16(aF[mi], bF[ni], acc[mi][ni]);

        cur ^= 1;
    }

    // ---- epilogue: bias (+ exact GELU), store ----
#pragma unroll
    for (int mi = 0; mi < 2; ++mi) {
#pragma unroll
        for (int ni = 0; ni < 4; ++ni) {
            const long n  = nb + wn * 64 + ni * 16 + ln;
            const float bv = bias[n];
#pragma unroll
            for (int r = 0; r < 8; ++r) {
                const long m = mb + wm * 32 + mi * 16 + hi * 8 + r;
                float v = acc[mi][ni][r] + bv;
                if (GELU) v = 0.5f * v * (1.0f + erff(v * 0.70710678118654752f));
                C[m * N + n] = (OutT)v;
            }
        }
    }
}

// ---------------------------------------------------------------------------
// Attention: one wave per (batch, head).  S=49 padded to 64.
//   S = tanh(Q Kt / 8)  (64x64, via 2 strips of 64x32, bf16 WMMA, K=64)
//   O = S @ V           (64x16, bf16 WMMA over seq-K, zero-padded)
// D-layout -> A-layout repack of tanh(S) goes through a per-wave LDS strip.
// ---------------------------------------------------------------------------
#define SSTR 40

__global__ __launch_bounds__(256) void attn_kernel(
    const __bf16* __restrict__ Q,    // [B][49][2048]
    const __bf16* __restrict__ Km,   // [B][49][2048]
    const __bf16* __restrict__ Vt,   // [32][16][64] (h,d,k) zero-padded
    __bf16*       __restrict__ Oa)   // [B][49][512]
{
    __shared__ __align__(16) __bf16 S_lds[8][64 * SSTR];

    const int t    = threadIdx.x;
    const int lane = t & 31;
    const int wave = t >> 5;
    const int ln   = lane & 15;
    const int hi   = lane >> 4;

    const int pair = blockIdx.x * 8 + wave;     // 0..8191
    const int b    = pair >> 5;
    const int h    = pair & 31;

    const __bf16* Qb = Q  + (long)b * SEQ * DQK + h * HQK;
    const __bf16* Kb = Km + (long)b * SEQ * DQK + h * HQK;
    const __bf16* Vh = Vt + h * (HV * 64);
    __bf16* Sw = &S_lds[wave][0];

    v8f Oacc[4] = {};

    for (int strip = 0; strip < 2; ++strip) {
        const int kbase = strip * 32;           // seq-k range of this strip
        v8f Sacc[4][2] = {};

#pragma unroll
        for (int kd = 0; kd < 2; ++kd) {        // d = kd*32 .. +31
            // B fragments from K matrix: lane n -> seq pos kq (predicated)
            v16bf bF[2];
#pragma unroll
            for (int nt = 0; nt < 2; ++nt) {
                const int kq = kbase + nt * 16 + ln;
                bf16x8 lo = zero8(), h8 = zero8();
                if (kq < SEQ) {
                    const __bf16* p = Kb + (long)kq * DQK + kd * 32 + hi * 8;
                    lo = *(const bf16x8*)p;
                    h8 = *(const bf16x8*)(p + 16);
                }
                bF[nt] = mk16(lo, h8);
            }
#pragma unroll
            for (int mt = 0; mt < 4; ++mt) {
                const int q = mt * 16 + ln;
                bf16x8 lo = zero8(), h8 = zero8();
                if (q < SEQ) {
                    const __bf16* p = Qb + (long)q * DQK + kd * 32 + hi * 8;
                    lo = *(const bf16x8*)p;
                    h8 = *(const bf16x8*)(p + 16);
                }
                const v16bf aF = mk16(lo, h8);
                Sacc[mt][0] = wmma_bf16(aF, bF[0], Sacc[mt][0]);
                Sacc[mt][1] = wmma_bf16(aF, bF[1], Sacc[mt][1]);
            }
        }

        // tanh(dot/8), D-layout -> LDS strip [64 rows][32 k-cols]
#pragma unroll
        for (int mt = 0; mt < 4; ++mt)
#pragma unroll
            for (int nt = 0; nt < 2; ++nt)
#pragma unroll
                for (int r = 0; r < 8; ++r) {
                    const int qr = mt * 16 + hi * 8 + r;
                    const int c  = nt * 16 + ln;
                    Sw[qr * SSTR + c] = (__bf16)tanhf(Sacc[mt][nt][r] * 0.125f);
                }
        __syncthreads();

        // V fragment: lane n = d, K = seq (contiguous in packed table)
        v16bf vF;
        {
            const __bf16* p = Vh + ln * 64 + kbase + hi * 8;
            vF = mk16(*(const bf16x8*)p, *(const bf16x8*)(p + 16));
        }
        // O += S_strip(64x32) @ V_strip(32x16)
#pragma unroll
        for (int mt = 0; mt < 4; ++mt) {
            const int row = mt * 16 + ln;
            const int off = hi * 8;
            const v16bf aF = mk16(*(const bf16x8*)(Sw + row * SSTR + off),
                                  *(const bf16x8*)(Sw + row * SSTR + 16 + off));
            Oacc[mt] = wmma_bf16(aF, vF, Oacc[mt]);
        }
        __syncthreads();
    }

    // store valid rows: out[b][q][h*16 + d]
#pragma unroll
    for (int mt = 0; mt < 4; ++mt)
#pragma unroll
        for (int r = 0; r < 8; ++r) {
            const int q = mt * 16 + hi * 8 + r;
            if (q < SEQ)
                Oa[((long)b * SEQ + q) * DMODEL + h * HV + ln] = (__bf16)Oacc[mt][r];
        }
}

// ---------------------------------------------------------------------------
// Host-side pipeline
// ---------------------------------------------------------------------------
extern "C" void kernel_launch(void* const* d_in, const int* in_sizes, int n_in,
                              void* d_out, int out_size, void* d_ws, size_t ws_size,
                              hipStream_t stream) {
    const float* x    = (const float*)d_in[0];
    const float* Wq1  = (const float*)d_in[1];
    const float* bq1  = (const float*)d_in[2];
    const float* Wq2  = (const float*)d_in[3];
    const float* bq2  = (const float*)d_in[4];
    const float* Wk1  = (const float*)d_in[5];
    const float* bk1  = (const float*)d_in[6];
    const float* Wk2  = (const float*)d_in[7];
    const float* bk2  = (const float*)d_in[8];
    const float* tok  = (const float*)d_in[9];
    const float* Wp   = (const float*)d_in[10];
    const float* bp   = (const float*)d_in[11];

    char* ws = (char*)d_ws;
    size_t off = 0;
    auto alloc = [&](size_t elems) -> __bf16* {
        off = (off + 255) & ~(size_t)255;
        __bf16* p = (__bf16*)(ws + off);
        off += elems * sizeof(__bf16);
        return p;
    };

    const long nX  = (long)MTOT * DMODEL;     // 6,422,528
    const long nQK = (long)MTOT * DQK;        // 25,690,112

    __bf16* xb    = alloc(nX);
    __bf16* Wq1t  = alloc((long)DMODEL * DMODEL);
    __bf16* Wq2t  = alloc((long)DQK * DMODEL);
    __bf16* Wk1t  = alloc((long)DMODEL * DMODEL);
    __bf16* Wk2t  = alloc((long)DQK * DMODEL);
    __bf16* Wpt   = alloc((long)DMODEL * DMODEL);
    __bf16* tokvt = alloc(NHEADS * HV * 64);
    __bf16* Hbuf  = alloc(nX);                // reused for q-hidden then k-hidden
    __bf16* Qb    = alloc(nQK);
    __bf16* Kb2   = alloc(nQK);
    __bf16* attnb = alloc(nX);
    (void)ws_size; (void)in_sizes; (void)n_in; (void)out_size;

    // conversions / packing
    cvt_bf16<<<(nX + 255) / 256, 256, 0, stream>>>(x, xb, (int)nX);
    cvt_t_bf16<<<(DMODEL * DMODEL + 255) / 256, 256, 0, stream>>>(Wq1, Wq1t, DMODEL, DMODEL);
    cvt_t_bf16<<<(DQK * DMODEL + 255) / 256, 256, 0, stream>>>(Wq2, Wq2t, DQK, DMODEL);
    cvt_t_bf16<<<(DMODEL * DMODEL + 255) / 256, 256, 0, stream>>>(Wk1, Wk1t, DMODEL, DMODEL);
    cvt_t_bf16<<<(DQK * DMODEL + 255) / 256, 256, 0, stream>>>(Wk2, Wk2t, DQK, DMODEL);
    cvt_t_bf16<<<(DMODEL * DMODEL + 255) / 256, 256, 0, stream>>>(Wp, Wpt, DMODEL, DMODEL);
    pack_tokens<<<128, 256, 0, stream>>>(tok, tokvt);

    const dim3 blk(256);
    const dim3 g512(MTOT / BM, DMODEL / BN);   // 98 x 4
    const dim3 g2048(MTOT / BM, DQK / BN);     // 98 x 16

    // q = gelu(x Wq1 + bq1) Wq2 + bq2
    gemm_bt<true,  __bf16><<<g512,  blk, 0, stream>>>(xb,   Wq1t, bq1, Hbuf, MTOT, DMODEL, DMODEL);
    gemm_bt<false, __bf16><<<g2048, blk, 0, stream>>>(Hbuf, Wq2t, bq2, Qb,   MTOT, DQK,    DMODEL);
    // k = gelu(x Wk1 + bk1) Wk2 + bk2   (Hbuf reused)
    gemm_bt<true,  __bf16><<<g512,  blk, 0, stream>>>(xb,   Wk1t, bk1, Hbuf, MTOT, DMODEL, DMODEL);
    gemm_bt<false, __bf16><<<g2048, blk, 0, stream>>>(Hbuf, Wk2t, bk2, Kb2,  MTOT, DQK,    DMODEL);

    // attention: 8192 (b,h) pairs, 8 waves/block
    attn_kernel<<<BATCH * NHEADS / 8, blk, 0, stream>>>(Qb, Kb2, tokvt, attnb);

    // final projection -> f32 output
    gemm_bt<false, float><<<g512, blk, 0, stream>>>(attnb, Wpt, bp, (float*)d_out,
                                                    MTOT, DMODEL, DMODEL);
}
